// Gate_61272003444757
// MI455X (gfx1250) — matrix operand order repaired and checked by reading, not compiled, not run on hardware
//
#include <hip/hip_runtime.h>
#include <hip/hip_bf16.h>
#include <stdint.h>

// Problem sizes (from reference)
#define B_   8
#define T_   2048
#define H_   64
#define M_   32
#define S_   20
#define E_   3
#define IN_  32
#define EPSV 1e-8f

typedef __attribute__((ext_vector_type(16))) _Float16 v16h;
typedef __attribute__((ext_vector_type(8)))  float    v8f;

union Frag16 { v16h h; uint32_t u[8]; };

// A-matrix (16-bit, 16x32) DWORD index for VGPR v, lane group grp (lane>>4):
// V0..3: K = 2v + 8*grp ; V4..7: K = 16 + 2(v-4) + 8*grp  (ISA 7.12.2)
__device__ __forceinline__ int hpA(int v, int grp) { return v + 4*grp + ((v >> 2) << 2); }
// B-matrix (16-bit, 32x16) DWORD index: same K order as A (V0..3: K = 2v + 8*grp ; V4..7: K = 16 + 2(v-4) + 8*grp)
__device__ __forceinline__ int hpB(int v, int grp) { return v + 4*grp + ((v >> 2) << 2); }

__device__ __forceinline__ uint32_t pack2h(_Float16 a, _Float16 b) {
    union { _Float16 h[2]; uint32_t u; } x;
    x.h[0] = a; x.h[1] = b; return x.u;
}
// LDS byte offset of a __shared__ object: generic LDS pointers are
// {aperture_hi32, lds_offset32}, so the low 32 bits are the DS offset.
__device__ __forceinline__ uint32_t lds_off(const void* p) {
    return (uint32_t)(uintptr_t)p;
}
__device__ __forceinline__ void async_copy_b128(uint32_t dst_lds, const void* src) {
    unsigned long long ga = (unsigned long long)(uintptr_t)src;
    asm volatile("global_load_async_to_lds_b128 %0, %1, off"
                 :: "v"(dst_lds), "v"(ga) : "memory");
}
__device__ __forceinline__ void wait_async0() {
    asm volatile("s_wait_asynccnt 0x0" ::: "memory");
}

// ---------------------------------------------------------------------------
// Kernel 1: memories[b,t,:] = softmax((x@input_query)@memory^T) @ memory
// ---------------------------------------------------------------------------
__global__ __launch_bounds__(256) void mem_read_kernel(
    const float* __restrict__ x, const float* __restrict__ memory,
    const float* __restrict__ iq, float* __restrict__ Mws)
{
    __shared__ float iqs[IN_ * M_];
    __shared__ float mems[S_ * M_];
    int tid = threadIdx.x;
    for (int i = tid; i < IN_ * M_; i += 256) iqs[i] = iq[i];
    for (int i = tid; i < S_ * M_;  i += 256) mems[i] = memory[i];
    __syncthreads();

    int idx = blockIdx.x * 256 + tid;            // token in [0, B*T)
    float xr[IN_];
#pragma unroll
    for (int i = 0; i < IN_; ++i) xr[i] = x[(size_t)idx * IN_ + i];

    float q[M_];
    for (int m = 0; m < M_; ++m) {
        float a = 0.f;
        for (int i = 0; i < IN_; ++i) a += xr[i] * iqs[i * M_ + m];
        q[m] = a;
    }
    float sc[S_]; float mx = -1e30f;
    for (int s = 0; s < S_; ++s) {
        float a = 0.f;
        for (int m = 0; m < M_; ++m) a += q[m] * mems[s * M_ + m];
        sc[s] = a; mx = fmaxf(mx, a);
    }
    float sum = 0.f;
    for (int s = 0; s < S_; ++s) { float e = __expf(sc[s] - mx); sc[s] = e; sum += e; }
    float inv = 1.0f / sum;
    for (int m = 0; m < M_; ++m) {
        float a = 0.f;
        for (int s = 0; s < S_; ++s) a += sc[s] * mems[s * M_ + m];
        Mws[(size_t)idx * M_ + m] = a * inv;
    }
}

// ---------------------------------------------------------------------------
// Kernel 2: Q/K/V = hidden @ W{q,k,v} via v_wmma_f32_16x16x32_f16, f16 out
// One block: (e,b, 128-row tile). 8 waves x 6 output tiles each (48 tiles).
// ---------------------------------------------------------------------------
__global__ __launch_bounds__(256) void qkv_kernel(
    const float* __restrict__ hidden, const float* __restrict__ Wq,
    const float* __restrict__ Wk, const float* __restrict__ Wv,
    _Float16* __restrict__ Qh, _Float16* __restrict__ Kh, _Float16* __restrict__ Vh)
{
    __shared__ _Float16 Hs[128 * H_];        // 16 KB, row-major [row][h]
    __shared__ _Float16 WT[3][M_ * H_];      // 12 KB, transposed [n][k]
    int tid = threadIdx.x;
    int eb = blockIdx.x >> 4;                // e*B + b
    int tile = blockIdx.x & 15;
    int rowbase = tile * 128;
    int e = eb >> 3;
    size_t ebT = (size_t)eb * T_;

    for (int i = tid; i < 128 * H_; i += 256)
        Hs[i] = (_Float16)hidden[(ebT + rowbase + (i >> 6)) * H_ + (i & 63)];
    for (int i = tid; i < H_ * M_; i += 256) {
        int k = i >> 5, n = i & 31;
        WT[0][n * H_ + k] = (_Float16)Wq[e * H_ * M_ + i];
        WT[1][n * H_ + k] = (_Float16)Wk[e * H_ * M_ + i];
        WT[2][n * H_ + k] = (_Float16)Wv[e * H_ * M_ + i];
    }
    __syncthreads();

    int wave = tid >> 5, lane = tid & 31, grp = lane >> 4, lo = lane & 15;
    const uint32_t* Hdw = (const uint32_t*)Hs;       // 32 dwords per row
    for (int ti = 0; ti < 6; ++ti) {
        int tt = wave * 6 + ti;
        int mat = tt / 16, loc = tt % 16;
        int rt = loc >> 1, ct = loc & 1;
        const uint32_t* Wdw = (const uint32_t*)WT[mat];
        Frag16 a, bfr; v8f c = {};
#pragma unroll
        for (int s = 0; s < 2; ++s) {                // K = 64 -> 2 WMMA steps
#pragma unroll
            for (int v = 0; v < 8; ++v) {
                a.u[v]   = Hdw[(rt * 16 + lo) * 32 + 16 * s + hpA(v, grp)];
                bfr.u[v] = Wdw[(ct * 16 + lo) * 32 + 16 * s + hpB(v, grp)];
            }
            c = __builtin_amdgcn_wmma_f32_16x16x32_f16(
                    false, a.h, false, bfr.h, (short)0, c, false, false);
        }
        _Float16* dst = (mat == 0) ? Qh : ((mat == 1) ? Kh : Vh);
#pragma unroll
        for (int j = 0; j < 8; ++j) {
            int r = rowbase + rt * 16 + j + 8 * grp;
            dst[(ebT + r) * M_ + ct * 16 + lo] = (_Float16)c[j];
        }
    }
}

// ---------------------------------------------------------------------------
// Kernel 3: per (e,b,16-row tile): S = Q K^T (full 16x2048 row block in LDS,
// exact softmax -> f16 P), out = P V via WMMA, cosine vs memories -> [B,T,E].
// K/Q tiles staged memory->LDS with global_load_async_to_lds_b128 (ASYNCcnt).
// ---------------------------------------------------------------------------
__global__ __launch_bounds__(256) void attn_cos_kernel(
    const _Float16* __restrict__ Qh, const _Float16* __restrict__ Kh,
    const _Float16* __restrict__ Vh, const float* __restrict__ Mws,
    float* __restrict__ out)
{
    __shared__ float    Srow[16 * T_];       // 128 KB f32 scores (exp, unnorm.)
    __shared__ _Float16 Ph[16 * T_];         //  64 KB f16 probabilities
    __shared__ _Float16 Qs[16 * M_];         //   1 KB Q tile
    __shared__ _Float16 Ks[8][16 * M_];      //   8 KB per-wave K tile
    __shared__ float    part[2 * 3 * 256];   //   6 KB PV partials
    __shared__ float    obuf[16 * M_];       //   2 KB final out tile
    int tid = threadIdx.x;
    int eb = blockIdx.x >> 7;                // 128 row-tiles per (e,b)
    int tile = blockIdx.x & 127;
    int rowbase = tile * 16;
    int e = eb >> 3, b = eb & 7;
    size_t ebT = (size_t)eb * T_;
    int wave = tid >> 5, lane = tid & 31, grp = lane >> 4, lo = lane & 15;

    // ---- async stage of the contiguous 1KB Q tile (waves 0-1) ----
    if (tid < 64)
        async_copy_b128(lds_off(&Qs[tid * 8]), Qh + (ebT + rowbase) * M_ + tid * 8);
    wait_async0();
    __syncthreads();

    // ---- Phase 1: S = Q K^T, 128 column tiles, 1 WMMA each ----
    Frag16 aq;
    {
        const uint32_t* Qdw = (const uint32_t*)Qs;   // 16 dwords per row
#pragma unroll
        for (int v = 0; v < 8; ++v) aq.u[v] = Qdw[lo * 16 + hpA(v, grp)];
    }
    for (int ct = wave; ct < 128; ct += 8) {
        // stage this wave's contiguous 1KB K tile into its LDS slot
        uint32_t kd = lds_off(&Ks[wave][lane * 16]);
        const _Float16* kg = Kh + (ebT + ct * 16) * M_ + lane * 16;
        async_copy_b128(kd,       kg);
        async_copy_b128(kd + 16u, kg + 8);
        if (ct + 8 < 128)
            __builtin_prefetch(Kh + (ebT + (ct + 8) * 16) * M_ + lane * 16, 0, 1);
        wait_async0();
        Frag16 bk;
        const uint32_t* Kdw = (const uint32_t*)Ks[wave];
#pragma unroll
        for (int v = 0; v < 8; ++v) bk.u[v] = Kdw[lo * 16 + hpB(v, grp)];
        v8f c = {};
        c = __builtin_amdgcn_wmma_f32_16x16x32_f16(
                false, aq.h, false, bk.h, (short)0, c, false, false);
#pragma unroll
        for (int j = 0; j < 8; ++j)
            Srow[(j + 8 * grp) * T_ + ct * 16 + lo] = c[j];
    }
    __syncthreads();

    // ---- Phase 2: exact softmax per row; write normalized P as f16 ----
    {
        int row = tid >> 4, j = tid & 15;
        float* R = Srow + row * T_;
        _Float16* P = Ph + row * T_;
        float mx = -1e30f;
        for (int k = j; k < T_; k += 16) mx = fmaxf(mx, R[k]);
        mx = fmaxf(mx, __shfl_xor(mx, 1)); mx = fmaxf(mx, __shfl_xor(mx, 2));
        mx = fmaxf(mx, __shfl_xor(mx, 4)); mx = fmaxf(mx, __shfl_xor(mx, 8));
        float sum = 0.f;
        for (int k = j; k < T_; k += 16) { float ev = __expf(R[k] - mx); R[k] = ev; sum += ev; }
        sum += __shfl_xor(sum, 1); sum += __shfl_xor(sum, 2);
        sum += __shfl_xor(sum, 4); sum += __shfl_xor(sum, 8);
        float inv = 1.0f / sum;
        for (int k = j; k < T_; k += 16) P[k] = (_Float16)(R[k] * inv);
    }
    __syncthreads();

    // ---- Phase 3: out = P V (16x32), K=2048 split over 4 waves per col tile ----
    int ch = wave & 1, q = wave >> 1;
    const uint32_t* Pdw = (const uint32_t*)Ph;       // 1024 dwords per row
    v8f c = {};
    for (int kb = q * 512; kb < q * 512 + 512; kb += 32) {
        Frag16 a, bv;
#pragma unroll
        for (int v = 0; v < 8; ++v) {
            a.u[v] = Pdw[lo * 1024 + (kb >> 1) + hpA(v, grp)];
            int kv = kb + 2 * hpB(v, grp);
            const _Float16* Vp = Vh + (ebT + kv) * M_ + ch * 16 + lo;
            bv.u[v] = pack2h(Vp[0], Vp[M_]);
        }
        c = __builtin_amdgcn_wmma_f32_16x16x32_f16(
                false, a.h, false, bv.h, (short)0, c, false, false);
    }
    if (q > 0) {
#pragma unroll
        for (int j = 0; j < 8; ++j)
            part[((ch * 3 + q - 1) * 32 + lane) * 8 + j] = c[j];
    }
    __syncthreads();
    if (q == 0) {
#pragma unroll
        for (int j = 0; j < 8; ++j) {
            for (int p = 0; p < 3; ++p)
                c[j] += part[((ch * 3 + p) * 32 + lane) * 8 + j];
            obuf[(j + 8 * grp) * M_ + ch * 16 + lo] = c[j];
        }
    }
    __syncthreads();

    // ---- Phase 4: cosine(memories[b,t], out[t]) -> out[b,t,e] ----
    {
        int row = tid >> 4, j = tid & 15;
        int t = rowbase + row;
        const float* mp = Mws + ((size_t)b * T_ + t) * M_;
        float o0 = obuf[row * M_ + j], o1 = obuf[row * M_ + j + 16];
        float m0 = mp[j], m1 = mp[j + 16];
        float dot = o0 * m0 + o1 * m1;
        float no  = o0 * o0 + o1 * o1;
        float nm  = m0 * m0 + m1 * m1;
        dot += __shfl_xor(dot, 1); no += __shfl_xor(no, 1); nm += __shfl_xor(nm, 1);
        dot += __shfl_xor(dot, 2); no += __shfl_xor(no, 2); nm += __shfl_xor(nm, 2);
        dot += __shfl_xor(dot, 4); no += __shfl_xor(no, 4); nm += __shfl_xor(nm, 4);
        dot += __shfl_xor(dot, 8); no += __shfl_xor(no, 8); nm += __shfl_xor(nm, 8);
        if (j == 0) {
            float na = fmaxf(sqrtf(nm), EPSV);
            float nb = fmaxf(sqrtf(no), EPSV);
            out[((size_t)b * T_ + t) * E_ + e] = dot / (na * nb);
        }
    }
}

// ---------------------------------------------------------------------------
extern "C" void kernel_launch(void* const* d_in, const int* in_sizes, int n_in,
                              void* d_out, int out_size, void* d_ws, size_t ws_size,
                              hipStream_t stream) {
    (void)in_sizes; (void)n_in; (void)out_size; (void)ws_size;
    const float* x      = (const float*)d_in[0];
    const float* hidden = (const float*)d_in[1];
    const float* memory = (const float*)d_in[2];
    const float* Wq     = (const float*)d_in[3];
    const float* Wk     = (const float*)d_in[4];
    const float* Wv     = (const float*)d_in[5];
    const float* iq     = (const float*)d_in[6];
    float* out = (float*)d_out;

    // Workspace layout (~11.5 MB total):
    //   [0, 2MB)           memories f32 [B,T,M]
    //   [2MB, +3MB each)   Qh, Kh, Vh  f16 [E,B,T,M]
    char* ws = (char*)d_ws;
    float*    Mws = (float*)ws;
    _Float16* Qh  = (_Float16*)(ws + (size_t)B_ * T_ * M_ * sizeof(float));
    _Float16* Kh  = Qh + (size_t)E_ * B_ * T_ * M_;
    _Float16* Vh  = Kh + (size_t)E_ * B_ * T_ * M_;

    mem_read_kernel<<<(B_ * T_) / 256, 256, 0, stream>>>(x, memory, iq, Mws);
    qkv_kernel<<<E_ * B_ * (T_ / 128), 256, 0, stream>>>(hidden, Wq, Wk, Wv, Qh, Kh, Vh);
    attn_cos_kernel<<<E_ * B_ * (T_ / 16), 256, 0, stream>>>(Qh, Kh, Vh, Mws, out);
}
